// SCSA_4818953306622
// MI455X (gfx1250) — matrix operand order repaired and checked by reading, not compiled, not run on hardware
//
#include <hip/hip_runtime.h>
#include <hip/hip_bf16.h>

// ---------------------------------------------------------------------------
// SCSA block for MI455X (gfx1250, wave32, WMMA).
//   x:(32,256,64,64) f32.  All GEMMs run on v_wmma_f32_16x16x32_bf16.
// ---------------------------------------------------------------------------

typedef __attribute__((ext_vector_type(8)))  __bf16 bf16x8;
typedef __attribute__((ext_vector_type(16))) __bf16 bf16x16;
typedef __attribute__((ext_vector_type(8)))  float  v8f;

__device__ __forceinline__ __bf16 f2bf(float f) {
    unsigned u = __builtin_bit_cast(unsigned, f);
    unsigned r = u + 0x7FFFu + ((u >> 16) & 1u);       // round-to-nearest-even
    unsigned short h = (unsigned short)(r >> 16);
    return __builtin_bit_cast(__bf16, h);
}

__device__ __forceinline__ bf16x16 load2x8(const __bf16* p0, const __bf16* p1) {
    bf16x8 lo = *(const bf16x8*)p0;
    bf16x8 hi = *(const bf16x8*)p1;
    return __builtin_shufflevector(lo, hi, 0,1,2,3,4,5,6,7,8,9,10,11,12,13,14,15);
}

// A-matrix fragment (16x32 bf16, this code's k order): lane L holds row M=L&15;
// lanes 0-15: K = kb*32 + 0..15; lanes 16-31: K = kb*32 + 16..31 (contiguous, matches B).
__device__ __forceinline__ bf16x16 loadA(const __bf16* rowPtr, int kb, int lane) {
    int hi = (lane >> 4) & 1;
    int k0 = kb * 32 + 16 * hi;
    return load2x8(rowPtr + k0, rowPtr + k0 + 8);
}

// B-matrix fragment (32x16 bf16): lane L holds col N=L&15; lanes 0-15 hold
// K=kb*32+0..15, lanes 16-31 K=kb*32+16..31 (contiguous 16 values).
__device__ __forceinline__ bf16x16 loadBlds(const __bf16* colBase, int colStride,
                                            int N, int kb, int lane) {
    int hi = (lane >> 4) & 1;
    return *(const bf16x16*)(colBase + (long)N * colStride + kb * 32 + 16 * hi);
}

// B fragment from pre-swizzled global buffer: frag[((kb*NT+nt)*32+lane)*16 + i]
__device__ __forceinline__ bf16x16 loadBfrag(const __bf16* frag, int ntiles,
                                             int kb, int nt, int lane) {
    return *(const bf16x16*)(frag + ((((long)kb * ntiles + nt) * 32 + lane) << 4));
}

__device__ __forceinline__ v8f wmma_bf16(bf16x16 a, bf16x16 b, v8f c) {
    return __builtin_amdgcn_wmma_f32_16x16x32_bf16(false, a, false, b,
                                                   (short)0, c, false, false);
}

__device__ __forceinline__ float sigm(float v) { return 1.f / (1.f + __expf(-v)); }

// ---------------------------------------------------------------------------
// K0: convert weights fp32 -> bf16 in WMMA B-fragment order.
// totals: qkv 256x768=196608, out 256x256=65536, ca_w1 256x64=16384, ca_w2 64x256=16384
// ---------------------------------------------------------------------------
__global__ __launch_bounds__(256) void wconv_kernel(
    const float* qkv_w, const float* out_w, const float* w1, const float* w2,
    __bf16* fqkv, __bf16* fout, __bf16* fw1, __bf16* fw2) {
    int e = blockIdx.x * 256 + threadIdx.x;     // 0 .. 294911
    const float* W; __bf16* F; int N;
    if (e < 196608)        { W = qkv_w; F = fqkv; N = 768; }
    else if (e < 262144)   { e -= 196608; W = out_w; F = fout; N = 256; }
    else if (e < 278528)   { e -= 262144; W = w1;   F = fw1;  N = 64;  }
    else                   { e -= 278528; W = w2;   F = fw2;  N = 256; }
    int NT = N >> 4;
    int tile   = e >> 9;            // 32 lanes * 16 elems
    int kb     = tile / NT;
    int nt     = tile % NT;
    int within = e & 511;
    int lane   = within >> 4;
    int i      = within & 15;
    int hi     = lane >> 4;
    int K      = kb * 32 + 16 * hi + i;
    int Nc     = nt * 16 + (lane & 15);
    F[e] = f2bf(W[(long)K * N + Nc]);
}

// ---------------------------------------------------------------------------
// K1: channel attention.  Block = 128 consecutive pixels (same image), 8 waves.
// x1 = x * sigmoid(relu(x @ W1 + b1) @ W2 + b2)
// LDS: Xbf 128x256 bf16 (64KB) + hidden 8x16x64 bf16 (16KB) = 80KB dynamic.
// ---------------------------------------------------------------------------
__global__ __launch_bounds__(256) void ca_kernel(
    const float* __restrict__ x, const float* ca_b1, const float* ca_b2,
    const __bf16* fw1, const __bf16* fw2, float* __restrict__ x1) {
    extern __shared__ char smem[];
    __bf16* Xbf = (__bf16*)smem;                    // 128*256
    __bf16* Hid = (__bf16*)(smem + 128 * 256 * 2);  // 8 * 16*64

    int t = threadIdx.x, lane = t & 31, wave = t >> 5;
    long pixBase = (long)blockIdx.x * 128;
    int  b   = (int)(pixBase >> 12);                // 4096 pixels/image
    int  sp0 = (int)(pixBase & 4095);
    const float* xb = x  + (long)b * 256 * 4096;
    float*       xo = x1 + (long)b * 256 * 4096;

    // stage 128 pixels x 256 channels as bf16 (coalesced over pixels)
    for (int chunk = 0; chunk < 128; ++chunk) {
        int e = chunk * 256 + t;
        int c = e >> 7, p = e & 127;
        Xbf[p * 256 + c] = f2bf(xb[(long)c * 4096 + sp0 + p]);
    }
    __syncthreads();

    int mrow0 = wave * 16;
    int hiL   = (lane >> 4) & 1;
    const __bf16* rowP = Xbf + (mrow0 + (lane & 15)) * 256;

    bf16x16 a[8];
    #pragma unroll
    for (int kb = 0; kb < 8; ++kb) a[kb] = loadA(rowP, kb, lane);

    // GEMM1: (16 x 256) @ (256 x 64) + relu -> hidden bf16
    __bf16* hid = Hid + wave * 16 * 64;
    #pragma unroll
    for (int nt = 0; nt < 4; ++nt) {
        v8f acc = {};
        #pragma unroll
        for (int kb = 0; kb < 8; ++kb)
            acc = wmma_bf16(a[kb], loadBfrag(fw1, 4, kb, nt, lane), acc);
        int n = nt * 16 + (lane & 15);
        float bias = ca_b1[n];
        #pragma unroll
        for (int r = 0; r < 8; ++r) {
            float v = acc[r] + bias;
            hid[(r + 8 * hiL) * 64 + n] = f2bf(v > 0.f ? v : 0.f);
        }
    }

    // GEMM2: (16 x 64) @ (64 x 256), sigmoid gate
    const __bf16* rowP2 = hid + (lane & 15) * 64;
    bf16x16 a0 = loadA(rowP2, 0, lane);
    bf16x16 a1 = loadA(rowP2, 1, lane);
    for (int nt = 0; nt < 16; ++nt) {
        v8f acc = {};
        acc = wmma_bf16(a0, loadBfrag(fw2, 16, 0, nt, lane), acc);
        acc = wmma_bf16(a1, loadBfrag(fw2, 16, 1, nt, lane), acc);
        int c = nt * 16 + (lane & 15);
        float bias = ca_b2[c];
        #pragma unroll
        for (int r = 0; r < 8; ++r) {
            int  p   = mrow0 + r + 8 * hiL;
            long idx = (long)c * 4096 + sp0 + p;
            xo[idx] = xb[idx] * sigm(acc[r] + bias);
        }
    }
}

// ---------------------------------------------------------------------------
// K2: windowed MHSA.  One block per 8x8 window (2048 blocks), 8 waves.
// LDS layout (dynamic, 155648 B):
//   QK  64x512 bf16  @0      (65536)
//   Vt 256x64  bf16  @65536  (32768)   V stored transposed: Vt[c][token]
//   Sc  64x64  f32   @98304  (16384)   per-head scores
//   P   64x64  bf16  @114688 (8192)    softmax probs
//   XO  64x256 bf16  @122880 (32768)   X tile, later reused as attention O
// ---------------------------------------------------------------------------
__global__ __launch_bounds__(256) void attn_kernel(
    const __bf16* fqkv, const __bf16* fout,
    const float* qkv_b, const float* out_b, float* __restrict__ x1) {
    extern __shared__ char smem[];
    __bf16* QK = (__bf16*)smem;
    __bf16* Vt = (__bf16*)(smem + 65536);
    float*  Sc = (float*)(smem + 98304);
    __bf16* P  = (__bf16*)(smem + 114688);
    __bf16* XO = (__bf16*)(smem + 122880);

    int t = threadIdx.x, lane = t & 31, wave = t >> 5;
    int hiL = (lane >> 4) & 1;
    int wid = blockIdx.x;
    int b = wid >> 6, win = wid & 63;
    int gy0 = (win >> 3) * 8, gx0 = (win & 7) * 8;
    float* xw = x1 + (long)b * 256 * 4096;

    // stage 64 tokens x 256 channels
    for (int it = 0; it < 64; ++it) {
        int e = it * 256 + t;
        int c = e >> 6, tok = e & 63;
        int ty = tok >> 3, tx = tok & 7;
        XO[tok * 256 + c] = f2bf(xw[((long)c * 64 + gy0 + ty) * 64 + gx0 + tx]);
    }
    __syncthreads();

    // ---- QKV GEMM: M=64, N=768, K=256 ----
    {
        int mt = wave & 3;
        const __bf16* rowP = XO + (mt * 16 + (lane & 15)) * 256;
        bf16x16 a[8];
        #pragma unroll
        for (int kb = 0; kb < 8; ++kb) a[kb] = loadA(rowP, kb, lane);
        for (int nt = (wave >> 2); nt < 48; nt += 2) {
            v8f acc = {};
            #pragma unroll
            for (int kb = 0; kb < 8; ++kb)
                acc = wmma_bf16(a[kb], loadBfrag(fqkv, 48, kb, nt, lane), acc);
            int n = nt * 16 + (lane & 15);
            float bias = qkv_b[n];
            #pragma unroll
            for (int r = 0; r < 8; ++r) {
                int tok = mt * 16 + r + 8 * hiL;
                float v = acc[r] + bias;
                if (n < 512) QK[tok * 512 + n] = f2bf(v);          // Q | K
                else         Vt[(n - 512) * 64 + tok] = f2bf(v);   // V transposed
            }
        }
    }
    __syncthreads();

    const float scale = 0.125f;   // hd=64 -> 1/sqrt(64)
    for (int h = 0; h < 4; ++h) {
        // scores = Q @ K^T  (16 tiles of 16x16, K=64)
        for (int idx = wave; idx < 16; idx += 8) {
            int mt = idx >> 2, nt = idx & 3;
            const __bf16* rowP  = QK + (mt * 16 + (lane & 15)) * 512 + h * 64;
            const __bf16* kbase = QK + 256 + h * 64;
            bf16x16 q0 = loadA(rowP, 0, lane);
            bf16x16 q1 = loadA(rowP, 1, lane);
            int n = nt * 16 + (lane & 15);
            v8f acc = {};
            acc = wmma_bf16(q0, loadBlds(kbase, 512, n, 0, lane), acc);
            acc = wmma_bf16(q1, loadBlds(kbase, 512, n, 1, lane), acc);
            #pragma unroll
            for (int r = 0; r < 8; ++r)
                Sc[(mt * 16 + r + 8 * hiL) * 64 + n] = acc[r] * scale;
        }
        __syncthreads();
        // softmax over rows (thread t < 64 owns one row)
        if (t < 64) {
            float mx = -3.4e38f;
            for (int j = 0; j < 64; ++j) mx = fmaxf(mx, Sc[t * 64 + j]);
            float s = 0.f;
            for (int j = 0; j < 64; ++j) s += __expf(Sc[t * 64 + j] - mx);
            float inv = 1.f / s;
            for (int j = 0; j < 64; ++j)
                P[t * 64 + j] = f2bf(__expf(Sc[t * 64 + j] - mx) * inv);
        }
        __syncthreads();
        // O = P @ V  (16 tiles, K=64 tokens)
        for (int idx = wave; idx < 16; idx += 8) {
            int mt = idx >> 2, nt = idx & 3;
            const __bf16* rowP = P + (mt * 16 + (lane & 15)) * 64;
            bf16x16 p0 = loadA(rowP, 0, lane);
            bf16x16 p1 = loadA(rowP, 1, lane);
            const __bf16* vb = Vt + h * 4096;
            int n = nt * 16 + (lane & 15);
            v8f acc = {};
            acc = wmma_bf16(p0, loadBlds(vb, 64, n, 0, lane), acc);
            acc = wmma_bf16(p1, loadBlds(vb, 64, n, 1, lane), acc);
            #pragma unroll
            for (int r = 0; r < 8; ++r)
                XO[(mt * 16 + r + 8 * hiL) * 256 + h * 64 + n] = f2bf(acc[r]);
        }
        __syncthreads();
    }

    // ---- out projection + residual: x1 += O @ Wout + b ----
    {
        int mt = wave & 3;
        const __bf16* rowP = XO + (mt * 16 + (lane & 15)) * 256;
        bf16x16 a[8];
        #pragma unroll
        for (int kb = 0; kb < 8; ++kb) a[kb] = loadA(rowP, kb, lane);
        for (int nt = (wave >> 2); nt < 16; nt += 2) {
            v8f acc = {};
            #pragma unroll
            for (int kb = 0; kb < 8; ++kb)
                acc = wmma_bf16(a[kb], loadBfrag(fout, 16, kb, nt, lane), acc);
            int c = nt * 16 + (lane & 15);
            float bias = out_b[c];
            #pragma unroll
            for (int r = 0; r < 8; ++r) {
                int tok = mt * 16 + r + 8 * hiL;
                int ty = tok >> 3, tx = tok & 7;
                long gi = ((long)c * 64 + gy0 + ty) * 64 + gx0 + tx;
                xw[gi] += acc[r] + bias;
            }
        }
    }
}

// ---------------------------------------------------------------------------
// K3: grouped conv1 (256->64, 7x7, groups=64) + bias + BN1 + relu.
// Channel shuffle folded in: shuffled in-channel 4g+j <- original j*64+g.
// ---------------------------------------------------------------------------
__global__ __launch_bounds__(256) void conv1_kernel(
    const float* __restrict__ x2, const float* __restrict__ w1, const float* b1,
    const float* g1, const float* bb1, const float* m1, const float* v1,
    float* __restrict__ y) {
    long o = (long)blockIdx.x * 256 + threadIdx.x;    // 32*64*64*64
    int ox = (int)(o & 63), oy = (int)((o >> 6) & 63);
    int g  = (int)((o >> 12) & 63), b = (int)(o >> 18);
    float acc = b1[g];
    for (int j = 0; j < 4; ++j) {
        const float* xin = x2 + ((long)b * 256 + j * 64 + g) * 4096;
        const float* wr  = w1 + (g * 4 + j) * 49;
        #pragma unroll
        for (int dy = 0; dy < 7; ++dy) {
            int iy = oy + dy - 3;
            if ((unsigned)iy > 63u) continue;
            #pragma unroll
            for (int dx = 0; dx < 7; ++dx) {
                int ix = ox + dx - 3;
                if ((unsigned)ix > 63u) continue;
                acc += xin[iy * 64 + ix] * wr[dy * 7 + dx];
            }
        }
    }
    float inv = g1[g] * __frsqrt_rn(v1[g] + 1e-5f);
    float val = acc * inv + (bb1[g] - m1[g] * inv);
    y[o] = val > 0.f ? val : 0.f;
}

// ---------------------------------------------------------------------------
// K4: grouped conv2 (64->256, 7x7, groups=64) + bias + BN2 + sigmoid, then
// out = x_shuffled * sigmoid(.)  with x_shuffled[c] = x2[(c&3)*64 + (c>>2)].
// ---------------------------------------------------------------------------
__global__ __launch_bounds__(256) void conv2_kernel(
    const float* __restrict__ ya, const float* __restrict__ x2,
    const float* __restrict__ w2, const float* b2c,
    const float* g2, const float* bb2, const float* m2, const float* v2,
    float* __restrict__ out) {
    long o = (long)blockIdx.x * 256 + threadIdx.x;    // 32*256*64*64
    int ox = (int)(o & 63), oy = (int)((o >> 6) & 63);
    int oc = (int)((o >> 12) & 255), b = (int)(o >> 20);
    const float* yin = ya + ((long)b * 64 + (oc >> 2)) * 4096;
    const float* wr  = w2 + oc * 49;
    float acc = b2c[oc];
    #pragma unroll
    for (int dy = 0; dy < 7; ++dy) {
        int iy = oy + dy - 3;
        if ((unsigned)iy > 63u) continue;
        #pragma unroll
        for (int dx = 0; dx < 7; ++dx) {
            int ix = ox + dx - 3;
            if ((unsigned)ix > 63u) continue;
            acc += yin[iy * 64 + ix] * wr[dy * 7 + dx];
        }
    }
    float inv = g2[oc] * __frsqrt_rn(v2[oc] + 1e-5f);
    float val = acc * inv + (bb2[oc] - m2[oc] * inv);
    int src = (oc & 3) * 64 + (oc >> 2);
    float xs = x2[(((long)b * 256 + src) * 64 + oy) * 64 + ox];
    out[o] = xs * sigm(val);
}

// ---------------------------------------------------------------------------
extern "C" void kernel_launch(void* const* d_in, const int* in_sizes, int n_in,
                              void* d_out, int out_size, void* d_ws, size_t ws_size,
                              hipStream_t stream) {
    (void)in_sizes; (void)n_in; (void)out_size; (void)ws_size;
    const float* x     = (const float*)d_in[0];
    const float* qkv_w = (const float*)d_in[1];
    const float* qkv_b = (const float*)d_in[2];
    const float* out_w = (const float*)d_in[3];
    const float* out_b = (const float*)d_in[4];
    const float* ca_w1 = (const float*)d_in[5];
    const float* ca_b1 = (const float*)d_in[6];
    const float* ca_w2 = (const float*)d_in[7];
    const float* ca_b2 = (const float*)d_in[8];
    const float* sa_w1 = (const float*)d_in[9];
    const float* sa_b1 = (const float*)d_in[10];
    const float* bn1_g = (const float*)d_in[11];
    const float* bn1_b = (const float*)d_in[12];
    const float* bn1_m = (const float*)d_in[13];
    const float* bn1_v = (const float*)d_in[14];
    const float* sa_w2 = (const float*)d_in[15];
    const float* sa_b2 = (const float*)d_in[16];
    const float* bn2_g = (const float*)d_in[17];
    const float* bn2_b = (const float*)d_in[18];
    const float* bn2_m = (const float*)d_in[19];
    const float* bn2_v = (const float*)d_in[20];

    char* ws = (char*)d_ws;
    float*  x1   = (float*)ws;                       // 134217728 B (gated x, then +attn)
    float*  yact = (float*)(ws + 134217728);         //  33554432 B
    __bf16* fqkv = (__bf16*)(ws + 167772160);        //    393216 B
    __bf16* fout = (__bf16*)(ws + 168165376);        //    131072 B
    __bf16* fw1  = (__bf16*)(ws + 168296448);        //     32768 B
    __bf16* fw2  = (__bf16*)(ws + 168329216);        //     32768 B

    wconv_kernel<<<1152, 256, 0, stream>>>(qkv_w, out_w, ca_w1, ca_w2,
                                           fqkv, fout, fw1, fw2);
    ca_kernel<<<1024, 256, 81920, stream>>>(x, ca_b1, ca_b2, fw1, fw2, x1);
    attn_kernel<<<2048, 256, 155648, stream>>>(fqkv, fout, qkv_b, out_b, x1);
    conv1_kernel<<<32768, 256, 0, stream>>>(x1, sa_w1, sa_b1,
                                            bn1_g, bn1_b, bn1_m, bn1_v, yact);
    conv2_kernel<<<131072, 256, 0, stream>>>(yact, x1, sa_w2, sa_b2,
                                             bn2_g, bn2_b, bn2_m, bn2_v,
                                             (float*)d_out);
}
